// JointModel_31327491457606
// MI455X (gfx1250) — matrix operand order repaired and checked by reading, not compiled, not run on hardware
//
#include <hip/hip_runtime.h>
#include <hip/hip_bf16.h>
#include <math.h>

// ---------------- problem constants ----------------
#define N_NODES  50000
#define N_EDGES  800000
#define F_NODE   1363
#define F_STRUCT 83
#define F_ESM    1280
#define H1       128
#define H2       64
#define LAYERS   4
#define EDGE_K   258      // 2*H1 + 1 + 1
#define EDGE_KP  288      // padded to multiple of 32
#define NODE_K   339      // H1 + H1 + F_STRUCT
#define NODE_KP  352
#define FF_H     256
#define DHEAD    192      // H1 + H2

typedef __attribute__((ext_vector_type(16))) __bf16 v16bf;
typedef __attribute__((ext_vector_type(8)))  float  v8f;

// ---------------- small helpers ----------------
__device__ __forceinline__ unsigned short f2bf(float f) {
  unsigned int x = __float_as_uint(f);
  unsigned int r = (x + 0x7FFFu + ((x >> 16) & 1u)) >> 16;   // round-to-nearest-even
  return (unsigned short)r;
}
__device__ __forceinline__ __bf16 bfu(unsigned short u) {
  union { unsigned short u; __bf16 b; } c; c.u = u; return c.b;
}
__device__ __forceinline__ float siluf(float x)    { return x / (1.f + __expf(-x)); }
__device__ __forceinline__ float sigmoidf_(float x){ return 1.f / (1.f + __expf(-x)); }

#define ZERO8 {0.f,0.f,0.f,0.f,0.f,0.f,0.f,0.f}

__device__ __forceinline__ v8f wmma_bf16(v16bf a, v16bf b, v8f c) {
  return __builtin_amdgcn_wmma_f32_16x16x32_bf16(false, a, false, b, (short)0, c, false, false);
}

// A fragment (16x32 bf16) from LDS tile (row-major, stride ldA):
//   lane<16 -> M=lane, K in {0..7}U{16..23}; lane>=16 -> +8.
// Each lane reads two contiguous 16B runs -> ds_load_b128 pairs.
__device__ __forceinline__ v16bf load_a(const unsigned short* As, int ldA, int kbase, int lane) {
  int m = lane & 15, hi = (lane >> 4) & 1;
  v16bf a;
#pragma unroll
  for (int i = 0; i < 16; ++i) {
    int k = kbase + (((i >> 1) & 3) * 2) + (i & 1) + hi * 8 + ((i >> 3) * 16);
    a[i] = bfu(As[m * ldA + k]);
  }
  return a;
}

// B fragment from fragment-major swizzled weights: each lane loads 32 contiguous
// bytes -> 2x global_load_b128. Fragment = 512 bf16 (32 lanes x 16 elems).
__device__ __forceinline__ v16bf load_bw(const unsigned short* Bg, int ktile, int ntile,
                                         int ntiles, int lane) {
  const unsigned short* p = Bg + ((size_t)(ktile * ntiles + ntile) * 512) + (size_t)lane * 16;
  return *(const v16bf*)p;
}

// ---------------- utility kernels ----------------
__global__ void cvt_swz_bf16_kernel(const float* __restrict__ src, unsigned short* __restrict__ dst,
                                    int Ksrc, int Kpad, int Ncols, int layers) {
  long long idx = (long long)blockIdx.x * blockDim.x + threadIdx.x;
  long long total = (long long)layers * Kpad * Ncols;
  if (idx >= total) return;
  int per = Kpad * Ncols;
  int l = (int)(idx / per);
  int rem = (int)(idx % per);
  int k = rem / Ncols, n = rem % Ncols;
  float v = (k < Ksrc) ? src[(long long)l * Ksrc * Ncols + (long long)k * Ncols + n] : 0.f;
  int ktile = k >> 5, kin = k & 31;
  int ntile = n >> 4, nin = n & 15;
  int hi = (kin >> 3) & 1, i = (kin & 7) + ((kin >> 4) << 3);
  int ntiles = Ncols >> 4;
  size_t off = ((size_t)l * (Kpad >> 5) * ntiles + (size_t)ktile * ntiles + ntile) * 512
             + (size_t)(hi * 16 + nin) * 16 + i;
  dst[off] = f2bf(v);
}

__global__ void zero_kernel(float* __restrict__ p, long long n) {
  long long i = (long long)blockIdx.x * blockDim.x + threadIdx.x;
  if (i < n) p[i] = 0.f;
}

__global__ void radial_kernel(const float* __restrict__ coords, const int* __restrict__ row,
                              const int* __restrict__ col, float* __restrict__ radial) {
  int e = blockIdx.x * blockDim.x + threadIdx.x;
  if (e >= N_EDGES) return;
  int r = row[e], c = col[e];
  float dx = coords[3*r]   - coords[3*c];
  float dy = coords[3*r+1] - coords[3*c+1];
  float dz = coords[3*r+2] - coords[3*c+2];
  radial[e] = dx*dx + dy*dy + dz*dz;
}

// h = h0 @ emb_w + emb_b  (h0 = node_attrs[:, :83])
__global__ void embed_kernel(const float* __restrict__ na, const float* __restrict__ w,
                             const float* __restrict__ b,
                             float* __restrict__ h, unsigned short* __restrict__ hbf) {
  int v = blockIdx.x;
  int j = threadIdx.x;
  const float* x = na + (long long)v * F_NODE;
  float acc = b[j];
  for (int k = 0; k < F_STRUCT; ++k) acc += x[k] * w[k * H1 + j];
  h[(long long)v * H1 + j]   = acc;
  hbf[(long long)v * H1 + j] = f2bf(acc);
}

// ---------------- edge message layer (32 edges / block, dual M-tiles / wave) ----------------
__global__ void __launch_bounds__(256)
edge_layer_kernel(const unsigned short* __restrict__ hbf,
                  const float* __restrict__ radial, const float* __restrict__ ea,
                  const int* __restrict__ row, const int* __restrict__ col,
                  const unsigned short* __restrict__ W1, const float* __restrict__ b1,
                  const unsigned short* __restrict__ W2, const float* __restrict__ b2,
                  const float* __restrict__ attw, const float* __restrict__ attb,
                  float* __restrict__ agg) {
  __shared__ unsigned short Ain[32][EDGE_KP];   // 18 KB
  __shared__ unsigned short T1[32][H1];         // 8 KB
  __shared__ float          Mm[32][H1];         // 16 KB
  __shared__ float          Satt[32];

  int tid = threadIdx.x, lane = tid & 31, wave = tid >> 5;
  int e0 = blockIdx.x * 32;

  // gather h[row] | h[col] in 8-byte units (32 rows x 32 units)
  for (int idx = tid; idx < 1024; idx += 256) {
    int r = idx >> 5, u = idx & 31;
    int e = e0 + r;
    const unsigned short* src = (u < 16)
        ? hbf + (size_t)row[e] * H1 + (size_t)u * 8
        : hbf + (size_t)col[e] * H1 + (size_t)(u - 16) * 8;
    *(unsigned long long*)&Ain[r][u * 8] = *(const unsigned long long*)src;
  }
  // tail columns 256..287: radial | ea | zero pad (32 rows x 32 cols)
  for (int idx = tid; idx < 1024; idx += 256) {
    int r = idx >> 5, cc = idx & 31;
    int e = e0 + r;
    unsigned short u = 0;
    if (cc == 0)      u = f2bf(radial[e]);
    else if (cc == 1) u = f2bf(ea[e]);
    Ain[r][256 + cc] = u;
  }
  __syncthreads();

  int n0 = wave * 16;
  int n  = lane & 15, hi = (lane >> 4) & 1;

  // GEMM1: (32 x 288) @ (288 x 128); B fragment shared by 2 M-tiles
  v8f cl = ZERO8, ch = ZERO8;
#pragma unroll
  for (int kk = 0; kk < EDGE_KP / 32; ++kk) {
    v16bf b  = load_bw(W1, kk, wave, H1 / 16, lane);
    v16bf a0 = load_a(&Ain[0][0],  EDGE_KP, kk * 32, lane);
    v16bf a1 = load_a(&Ain[16][0], EDGE_KP, kk * 32, lane);
    cl = wmma_bf16(a0, b, cl);
    ch = wmma_bf16(a1, b, ch);
  }
  {
    float bias = b1[n0 + n];
#pragma unroll
    for (int r8 = 0; r8 < 8; ++r8) {
      T1[r8 + hi * 8][n0 + n]      = f2bf(siluf(cl[r8] + bias));
      T1[16 + r8 + hi * 8][n0 + n] = f2bf(siluf(ch[r8] + bias));
    }
  }
  __syncthreads();

  // GEMM2: (32 x 128) @ (128 x 128)
  v8f dl = ZERO8, dh = ZERO8;
#pragma unroll
  for (int kk = 0; kk < H1 / 32; ++kk) {
    v16bf b  = load_bw(W2, kk, wave, H1 / 16, lane);
    v16bf a0 = load_a(&T1[0][0],  H1, kk * 32, lane);
    v16bf a1 = load_a(&T1[16][0], H1, kk * 32, lane);
    dl = wmma_bf16(a0, b, dl);
    dh = wmma_bf16(a1, b, dh);
  }
  {
    float bias = b2[n0 + n];
#pragma unroll
    for (int r8 = 0; r8 < 8; ++r8) {
      Mm[r8 + hi * 8][n0 + n]      = siluf(dl[r8] + bias);
      Mm[16 + r8 + hi * 8][n0 + n] = siluf(dh[r8] + bias);
    }
  }
  __syncthreads();

  // attention gate: 8 threads per row, shfl tree reduction
  {
    int r = tid >> 3, c = tid & 7;
    float p = 0.f;
#pragma unroll
    for (int t = 0; t < 16; ++t) {
      int j = c + t * 8;
      p += Mm[r][j] * attw[j];
    }
    p += __shfl_xor(p, 1, 32);
    p += __shfl_xor(p, 2, 32);
    p += __shfl_xor(p, 4, 32);
    if (c == 0) Satt[r] = sigmoidf_(p + attb[0]);
  }
  __syncthreads();

  // scatter-add into agg[row[e]]
  for (int idx = tid; idx < 32 * H1; idx += 256) {
    int r = idx >> 7, j = idx & 127;
    int e = e0 + r;
    unsafeAtomicAdd(&agg[(size_t)row[e] * H1 + j], Mm[r][j] * Satt[r]);
  }
}

// ---------------- node update layer ----------------
__global__ void __launch_bounds__(256)
node_layer_kernel(float* __restrict__ h, unsigned short* __restrict__ hbf,
                  const float* __restrict__ agg, const float* __restrict__ na,
                  const unsigned short* __restrict__ W1, const float* __restrict__ b1,
                  const unsigned short* __restrict__ W2, const float* __restrict__ b2) {
  __shared__ unsigned short Ain[16][NODE_KP];
  __shared__ unsigned short T1[16][H1];

  int tid = threadIdx.x, lane = tid & 31, wave = tid >> 5;
  int v0 = blockIdx.x * 16;

  // h part: 16 rows x 16 8-byte units
  {
    int r = tid >> 4, u = tid & 15;
    int v = v0 + r;
    *(unsigned long long*)&Ain[r][u * 8] =
        *(const unsigned long long*)(hbf + (size_t)v * H1 + (size_t)u * 8);
  }
  // agg part (f32 -> bf16), 2-float units: 16 rows x 64 units
  for (int idx = tid; idx < 1024; idx += 256) {
    int r = idx >> 6, u = idx & 63;
    const float* s = agg + (size_t)(v0 + r) * H1 + (size_t)u * 2;
    unsigned int pk = (unsigned int)f2bf(s[0]) | ((unsigned int)f2bf(s[1]) << 16);
    *(unsigned int*)&Ain[r][H1 + u * 2] = pk;
  }
  // h0 + zero pad: columns 256..351
  {
    int r = tid >> 4, cc = tid & 15;
    int v = v0 + r;
#pragma unroll
    for (int t = 0; t < 6; ++t) {
      int kk = cc + t * 16;   // 0..95
      unsigned short u = (kk < F_STRUCT) ? f2bf(na[(size_t)v * F_NODE + kk]) : (unsigned short)0;
      Ain[r][2 * H1 + kk] = u;
    }
  }
  __syncthreads();

  int n0 = wave * 16;
  int n  = lane & 15, hi = (lane >> 4) & 1;

  v8f c = ZERO8;
#pragma unroll
  for (int kk = 0; kk < NODE_KP / 32; ++kk) {
    v16bf a = load_a(&Ain[0][0], NODE_KP, kk * 32, lane);
    v16bf b = load_bw(W1, kk, wave, H1 / 16, lane);
    c = wmma_bf16(a, b, c);
  }
  {
    float bias = b1[n0 + n];
#pragma unroll
    for (int r8 = 0; r8 < 8; ++r8)
      T1[r8 + hi * 8][n0 + n] = f2bf(siluf(c[r8] + bias));
  }
  __syncthreads();

  v8f c2 = ZERO8;
#pragma unroll
  for (int kk = 0; kk < H1 / 32; ++kk) {
    v16bf a = load_a(&T1[0][0], H1, kk * 32, lane);
    v16bf b = load_bw(W2, kk, wave, H1 / 16, lane);
    c2 = wmma_bf16(a, b, c2);
  }
  {
    float bias = b2[n0 + n];
#pragma unroll
    for (int r8 = 0; r8 < 8; ++r8) {
      int v = v0 + r8 + hi * 8;
      float hn = h[(size_t)v * H1 + n0 + n] + c2[r8] + bias;   // residual
      h[(size_t)v * H1 + n0 + n]   = hn;
      hbf[(size_t)v * H1 + n0 + n] = f2bf(hn);
    }
  }
}

// ---------------- ESM FFNN head ----------------
__global__ void __launch_bounds__(256)
ffnn_kernel(const float* __restrict__ na,
            const unsigned short* __restrict__ W1, const float* __restrict__ b1,
            const unsigned short* __restrict__ W2, const float* __restrict__ b2,
            float* __restrict__ esm) {
  __shared__ unsigned short Ain[16][F_ESM];   // 40 KB
  __shared__ unsigned short T1[16][FF_H];     // 8 KB

  int tid = threadIdx.x, lane = tid & 31, wave = tid >> 5;
  int v0 = blockIdx.x * 16;
  int n  = lane & 15, hi = (lane >> 4) & 1;

  for (int r = 0; r < 16; ++r) {
    const float* s = na + (size_t)(v0 + r) * F_NODE + F_STRUCT;
    for (int k = tid; k < F_ESM; k += 256)
      Ain[r][k] = f2bf(s[k]);
  }
  __syncthreads();

  // GEMM1: (16 x 1280) @ (1280 x 256) -> ReLU ; A fragment shared by 2 N-tiles
  {
    int nt0 = wave, nt1 = wave + 8;
    v8f c0 = ZERO8, c1 = ZERO8;
#pragma unroll 4
    for (int kk = 0; kk < F_ESM / 32; ++kk) {
      v16bf a  = load_a(&Ain[0][0], F_ESM, kk * 32, lane);
      v16bf b0 = load_bw(W1, kk, nt0, FF_H / 16, lane);
      v16bf b1 = load_bw(W1, kk, nt1, FF_H / 16, lane);
      c0 = wmma_bf16(a, b0, c0);
      c1 = wmma_bf16(a, b1, c1);
    }
    float bias0 = b1[nt0 * 16 + n];
    float bias1 = b1[nt1 * 16 + n];
#pragma unroll
    for (int r8 = 0; r8 < 8; ++r8) {
      T1[r8 + hi * 8][nt0 * 16 + n] = f2bf(fmaxf(c0[r8] + bias0, 0.f));
      T1[r8 + hi * 8][nt1 * 16 + n] = f2bf(fmaxf(c1[r8] + bias1, 0.f));
    }
  }
  __syncthreads();

  // GEMM2: (16 x 256) @ (256 x 64) -> ReLU ; waves 0..3
  if (wave < 4) {
    int n0 = wave * 16;
    v8f c = ZERO8;
#pragma unroll
    for (int kk = 0; kk < FF_H / 32; ++kk) {
      v16bf a = load_a(&T1[0][0], FF_H, kk * 32, lane);
      v16bf b = load_bw(W2, kk, wave, H2 / 16, lane);
      c = wmma_bf16(a, b, c);
    }
    float bias = b2[n0 + n];
#pragma unroll
    for (int r8 = 0; r8 < 8; ++r8)
      esm[(size_t)(v0 + r8 + hi * 8) * H2 + n0 + n] = fmaxf(c[r8] + bias, 0.f);
  }
}

// ---------------- decoder + final head (fused) ----------------
__global__ void __launch_bounds__(256)
head_kernel(const float* __restrict__ h, const float* __restrict__ esm,
            const unsigned short* __restrict__ Wd1, const float* __restrict__ bd1,
            const unsigned short* __restrict__ Wd2, const float* __restrict__ bd2,
            const unsigned short* __restrict__ Wl1, const float* __restrict__ bl1,
            const float* __restrict__ wl2, const float* __restrict__ bl2,
            float* __restrict__ out) {
  __shared__ unsigned short Hs[16][H1];
  __shared__ unsigned short T1[16][H1];
  __shared__ unsigned short Cat[16][DHEAD];
  __shared__ float          Ls[16][DHEAD];

  int tid = threadIdx.x, lane = tid & 31, wave = tid >> 5;
  int v0 = blockIdx.x * 16;
  int n  = lane & 15, hi = (lane >> 4) & 1;

  // stage h (f32 -> bf16), 2-float units
  for (int idx = tid; idx < 1024; idx += 256) {
    int r = idx >> 6, u = idx & 63;
    const float* s = h + (size_t)(v0 + r) * H1 + (size_t)u * 2;
    unsigned int pk = (unsigned int)f2bf(s[0]) | ((unsigned int)f2bf(s[1]) << 16);
    *(unsigned int*)&Hs[r][u * 2] = pk;
  }
  // stage esm into Cat[:, 128:192]
  for (int idx = tid; idx < 512; idx += 256) {
    int r = idx >> 5, u = idx & 31;
    const float* s = esm + (size_t)(v0 + r) * H2 + (size_t)u * 2;
    unsigned int pk = (unsigned int)f2bf(s[0]) | ((unsigned int)f2bf(s[1]) << 16);
    *(unsigned int*)&Cat[r][H1 + u * 2] = pk;
  }
  __syncthreads();

  int n0 = wave * 16;

  // decoder GEMM1 + SiLU
  v8f c = ZERO8;
#pragma unroll
  for (int kk = 0; kk < H1 / 32; ++kk) {
    v16bf a = load_a(&Hs[0][0], H1, kk * 32, lane);
    v16bf b = load_bw(Wd1, kk, wave, H1 / 16, lane);
    c = wmma_bf16(a, b, c);
  }
  {
    float bias = bd1[n0 + n];
#pragma unroll
    for (int r8 = 0; r8 < 8; ++r8)
      T1[r8 + hi * 8][n0 + n] = f2bf(siluf(c[r8] + bias));
  }
  __syncthreads();

  // decoder GEMM2 -> Cat[:, :128]
  v8f c2 = ZERO8;
#pragma unroll
  for (int kk = 0; kk < H1 / 32; ++kk) {
    v16bf a = load_a(&T1[0][0], H1, kk * 32, lane);
    v16bf b = load_bw(Wd2, kk, wave, H1 / 16, lane);
    c2 = wmma_bf16(a, b, c2);
  }
  {
    float bias = bd2[n0 + n];
#pragma unroll
    for (int r8 = 0; r8 < 8; ++r8)
      Cat[r8 + hi * 8][n0 + n] = f2bf(c2[r8] + bias);
  }
  __syncthreads();

  // last GEMM1: (16 x 192) @ (192 x 192) + SiLU ; 12 tiles over 8 waves
  {
    v8f c3 = ZERO8;
#pragma unroll
    for (int kk = 0; kk < DHEAD / 32; ++kk) {
      v16bf a = load_a(&Cat[0][0], DHEAD, kk * 32, lane);
      v16bf b = load_bw(Wl1, kk, wave, DHEAD / 16, lane);
      c3 = wmma_bf16(a, b, c3);
    }
    float bias = bl1[n0 + n];
#pragma unroll
    for (int r8 = 0; r8 < 8; ++r8)
      Ls[r8 + hi * 8][n0 + n] = siluf(c3[r8] + bias);
  }
  if (wave < 4) {
    int nt = 8 + wave;
    int n1 = nt * 16;
    v8f c4 = ZERO8;
#pragma unroll
    for (int kk = 0; kk < DHEAD / 32; ++kk) {
      v16bf a = load_a(&Cat[0][0], DHEAD, kk * 32, lane);
      v16bf b = load_bw(Wl1, kk, nt, DHEAD / 16, lane);
      c4 = wmma_bf16(a, b, c4);
    }
    float bias = bl1[n1 + n];
#pragma unroll
    for (int r8 = 0; r8 < 8; ++r8)
      Ls[r8 + hi * 8][n1 + n] = siluf(c4[r8] + bias);
  }
  __syncthreads();

  // final: sigmoid(Ls @ last_w2 + b) ; 16 threads per row + shfl reduction
  {
    int r = tid >> 4, cidx = tid & 15;
    float p = 0.f;
#pragma unroll
    for (int t = 0; t < 12; ++t) {
      int j = cidx + t * 16;
      p += Ls[r][j] * wl2[j];
    }
    p += __shfl_xor(p, 1, 32);
    p += __shfl_xor(p, 2, 32);
    p += __shfl_xor(p, 4, 32);
    p += __shfl_xor(p, 8, 32);
    if (cidx == 0) out[v0 + r] = sigmoidf_(p + bl2[0]);
  }
}

// ---------------- host launch ----------------
extern "C" void kernel_launch(void* const* d_in, const int* in_sizes, int n_in,
                              void* d_out, int out_size, void* d_ws, size_t ws_size,
                              hipStream_t stream) {
  (void)in_sizes; (void)n_in; (void)out_size; (void)ws_size;

  const float* node_attrs = (const float*)d_in[0];
  const float* coords     = (const float*)d_in[1];
  const int*   edge_index = (const int*)  d_in[2];
  const float* edge_attrs = (const float*)d_in[3];
  const float* emb_w   = (const float*)d_in[4];
  const float* emb_b   = (const float*)d_in[5];
  const float* edge_w1 = (const float*)d_in[6];
  const float* edge_b1 = (const float*)d_in[7];
  const float* edge_w2 = (const float*)d_in[8];
  const float* edge_b2 = (const float*)d_in[9];
  const float* att_w   = (const float*)d_in[10];
  const float* att_b   = (const float*)d_in[11];
  const float* node_w1 = (const float*)d_in[12];
  const float* node_b1 = (const float*)d_in[13];
  const float* node_w2 = (const float*)d_in[14];
  const float* node_b2 = (const float*)d_in[15];
  const float* dec_w1  = (const float*)d_in[16];
  const float* dec_b1  = (const float*)d_in[17];
  const float* dec_w2  = (const float*)d_in[18];
  const float* dec_b2  = (const float*)d_in[19];
  const float* ffnn_w1 = (const float*)d_in[20];
  const float* ffnn_b1 = (const float*)d_in[21];
  const float* ffnn_w2 = (const float*)d_in[22];
  const float* ffnn_b2 = (const float*)d_in[23];
  const float* last_w1 = (const float*)d_in[24];
  const float* last_b1 = (const float*)d_in[25];
  const float* last_w2 = (const float*)d_in[26];
  const float* last_b2 = (const float*)d_in[27];

  const int* row = edge_index;
  const int* col = edge_index + N_EDGES;

  char* ws = (char*)d_ws;
  size_t off = 0;
  auto carve = [&](size_t bytes) -> void* {
    void* p = ws + off;
    off = (off + bytes + 255) & ~(size_t)255;
    return p;
  };
  float*          h      = (float*)         carve((size_t)N_NODES * H1 * 4);
  unsigned short* hbf    = (unsigned short*)carve((size_t)N_NODES * H1 * 2);
  float*          agg    = (float*)         carve((size_t)N_NODES * H1 * 4);
  float*          radial = (float*)         carve((size_t)N_EDGES * 4);
  float*          esm    = (float*)         carve((size_t)N_NODES * H2 * 4);
  unsigned short* wE1 = (unsigned short*)carve((size_t)LAYERS * EDGE_KP * H1 * 2);
  unsigned short* wE2 = (unsigned short*)carve((size_t)LAYERS * H1 * H1 * 2);
  unsigned short* wN1 = (unsigned short*)carve((size_t)LAYERS * NODE_KP * H1 * 2);
  unsigned short* wN2 = (unsigned short*)carve((size_t)LAYERS * H1 * H1 * 2);
  unsigned short* wD1 = (unsigned short*)carve((size_t)H1 * H1 * 2);
  unsigned short* wD2 = (unsigned short*)carve((size_t)H1 * H1 * 2);
  unsigned short* wF1 = (unsigned short*)carve((size_t)F_ESM * FF_H * 2);
  unsigned short* wF2 = (unsigned short*)carve((size_t)FF_H * H2 * 2);
  unsigned short* wL1 = (unsigned short*)carve((size_t)DHEAD * DHEAD * 2);

  auto cvt = [&](const float* s, unsigned short* d, int Ks, int Kp, int Nc, int L) {
    long long total = (long long)L * Kp * Nc;
    int grid = (int)((total + 255) / 256);
    cvt_swz_bf16_kernel<<<grid, 256, 0, stream>>>(s, d, Ks, Kp, Nc, L);
  };
  cvt(edge_w1, wE1, EDGE_K, EDGE_KP, H1, LAYERS);
  cvt(edge_w2, wE2, H1, H1, H1, LAYERS);
  cvt(node_w1, wN1, NODE_K, NODE_KP, H1, LAYERS);
  cvt(node_w2, wN2, H1, H1, H1, LAYERS);
  cvt(dec_w1,  wD1, H1, H1, H1, 1);
  cvt(dec_w2,  wD2, H1, H1, H1, 1);
  cvt(ffnn_w1, wF1, F_ESM, F_ESM, FF_H, 1);
  cvt(ffnn_w2, wF2, FF_H, FF_H, H2, 1);
  cvt(last_w1, wL1, DHEAD, DHEAD, DHEAD, 1);

  radial_kernel<<<N_EDGES / 256, 256, 0, stream>>>(coords, row, col, radial);
  embed_kernel<<<N_NODES, H1, 0, stream>>>(node_attrs, emb_w, emb_b, h, hbf);

  for (int i = 0; i < LAYERS; ++i) {
    long long aggN = (long long)N_NODES * H1;
    zero_kernel<<<(int)((aggN + 255) / 256), 256, 0, stream>>>(agg, aggN);
    edge_layer_kernel<<<N_EDGES / 32, 256, 0, stream>>>(
        hbf, radial, edge_attrs, row, col,
        wE1 + (size_t)i * EDGE_KP * H1, edge_b1 + i * H1,
        wE2 + (size_t)i * H1 * H1,      edge_b2 + i * H1,
        att_w + i * H1, att_b + i, agg);
    node_layer_kernel<<<N_NODES / 16, 256, 0, stream>>>(
        h, hbf, agg, node_attrs,
        wN1 + (size_t)i * NODE_KP * H1, node_b1 + i * H1,
        wN2 + (size_t)i * H1 * H1,      node_b2 + i * H1);
  }

  ffnn_kernel<<<N_NODES / 16, 256, 0, stream>>>(node_attrs, wF1, ffnn_b1, wF2, ffnn_b2, esm);
  head_kernel<<<N_NODES / 16, 256, 0, stream>>>(h, esm, wD1, dec_b1, wD2, dec_b2,
                                                wL1, last_b1, last_w2, last_b2,
                                                (float*)d_out);
}
